// SelfAttention_76312978915582
// MI455X (gfx1250) — matrix operand order repaired and checked by reading, not compiled, not run on hardware
//
#include <hip/hip_runtime.h>

typedef float v8f __attribute__((ext_vector_type(8)));
typedef __bf16 v16bf __attribute__((ext_vector_type(16)));
typedef unsigned short u16;
typedef unsigned int u32;
typedef int i32x4 __attribute__((vector_size(16)));

namespace {
constexpr int kB = 4;
constexpr int kC = 256;
constexpr int kN = 4096;            // 64*64 pixels
constexpr float kScale = 0.0625f;   // C^-0.5 = 1/16
}

#define AS1 __attribute__((address_space(1)))
#define AS3 __attribute__((address_space(3)))

#if defined(__has_builtin)
#  if __has_builtin(__builtin_amdgcn_global_load_async_to_lds_b128)
#    define USE_ASYNC_LDS 1
#  endif
#endif
#ifndef USE_ASYNC_LDS
#  define USE_ASYNC_LDS 0
#endif

// Copy 16 bytes global -> LDS (per lane). Async path uses the CDNA5
// GLOBAL_LOAD_ASYNC_TO_LDS_B128 (ASYNCcnt-tracked); fallback is load+ds_store.
__device__ __forceinline__ void cp16_to_lds(u16* dst_lds, const u16* src_g) {
#if USE_ASYNC_LDS
  __builtin_amdgcn_global_load_async_to_lds_b128(
      (AS1 i32x4*)src_g, (AS3 i32x4*)dst_lds, 0, 0);
#else
  *(uint4*)dst_lds = *(const uint4*)src_g;
#endif
}

__device__ __forceinline__ void wait_lds_copies() {
#if USE_ASYNC_LDS
#  if __has_builtin(__builtin_amdgcn_s_wait_asynccnt)
  __builtin_amdgcn_s_wait_asynccnt(0);
#  else
  asm volatile("s_wait_asynccnt 0x0" ::: "memory");
#  endif
#endif
}

__device__ __forceinline__ u16 f2bf(float f) {
  u32 u = __builtin_bit_cast(u32, f);
  u = (u + 0x7fffu + ((u >> 16) & 1u)) >> 16;   // round-to-nearest-even
  return (u16)u;
}
__device__ __forceinline__ float bf2f(u16 h) {
  u32 u = ((u32)h) << 16;
  return __builtin_bit_cast(float, u);
}

struct Frag32 { uint4 lo, hi; };   // 32 bytes = 16 bf16

// A-fragment (16x32 bf16). ISA layout: lane L (m=L&15, h=L>>4):
//   v0..3 hold K = 8h+0..7, v4..7 hold K = 16+8h+0..7  -> two contiguous 16B loads.
__device__ __forceinline__ v16bf load_afrag(const u16* tile, int ld, int cb, int m, int h) {
  Frag32 f;
  const u16* p = tile + (size_t)m * ld + cb + 8 * h;
  f.lo = *(const uint4*)p;
  f.hi = *(const uint4*)(p + 16);
  return __builtin_bit_cast(v16bf, f);
}

// B-fragment (32x16 bf16) sourced from Bt row-major [n][k]:
//   lane L (n=L&15, h=L>>4) reads Bt[n][cb+8h .. +7] and Bt[n][cb+16+8h .. +7] -> two 16B loads.
__device__ __forceinline__ v16bf load_bfrag(const u16* bt, int ld, int cb, int n, int h) {
  Frag32 f;
  const u16* p = bt + (size_t)n * ld + cb + 8 * h;
  f.lo = *(const uint4*)p;
  f.hi = *(const uint4*)(p + 16);
  return __builtin_bit_cast(v16bf, f);
}

__device__ __forceinline__ v8f wmma_bf16(v16bf a, v16bf b, v8f c) {
  return __builtin_amdgcn_wmma_f32_16x16x32_bf16(false, a, false, b, (short)0, c, false, false);
}

// ---------------------------------------------------------------- weights -> bf16
__global__ void k_convert_bf16(const float* __restrict__ src, u16* __restrict__ dst, int count) {
  int i = blockIdx.x * blockDim.x + threadIdx.x;
  if (i < count) dst[i] = f2bf(src[i]);
}

// ---------------------------------------------------------------- x [B,C,N] f32 -> Xr [B,N,C] bf16
__global__ void k_pack_x(const float* __restrict__ x, u16* __restrict__ Xr) {
  __shared__ float t[32][33];
  int b = blockIdx.z;
  int n0 = blockIdx.x * 32, c0 = blockIdx.y * 32;
  const float* xb = x + (size_t)b * kC * kN;
  u16* Xb = Xr + (size_t)b * kN * kC;
#pragma unroll
  for (int k = 0; k < 4; ++k) {
    int c = c0 + threadIdx.y + k * 8;
    t[threadIdx.y + k * 8][threadIdx.x] = xb[(size_t)c * kN + n0 + threadIdx.x];
  }
  __syncthreads();
#pragma unroll
  for (int k = 0; k < 4; ++k) {
    int n = n0 + threadIdx.y + k * 8;
    Xb[(size_t)n * kC + c0 + threadIdx.x] = f2bf(t[threadIdx.x][threadIdx.y + k * 8]);
  }
}

// ---------------------------------------------------------------- Vr [B,N,C] -> Vt [B,C,N] (bf16)
__global__ void k_transpose_v(const u16* __restrict__ Vr, u16* __restrict__ Vt) {
  __shared__ u16 t[32][33];
  int b = blockIdx.z;
  int n0 = blockIdx.x * 32, c0 = blockIdx.y * 32;
  const u16* src = Vr + (size_t)b * kN * kC;
  u16* dst = Vt + (size_t)b * kC * kN;
#pragma unroll
  for (int k = 0; k < 4; ++k) {
    int n = n0 + threadIdx.y + k * 8;
    t[threadIdx.y + k * 8][threadIdx.x] = src[(size_t)n * kC + c0 + threadIdx.x];
  }
  __syncthreads();
#pragma unroll
  for (int k = 0; k < 4; ++k) {
    int c = c0 + threadIdx.y + k * 8;
    dst[(size_t)c * kN + n0 + threadIdx.x] = t[threadIdx.x][threadIdx.y + k * 8];
  }
}

// ---------------------------------------------------------------- fused QKV projection
// Qr/Kr/Vr[b][i][c] = sum_c' Xr[b][i][c'] * W[c][c'] + bias[c]   (Bt == W row-major)
__global__ __launch_bounds__(128) void k_qkv(
    const u16* __restrict__ Xr,
    const u16* __restrict__ Wqb, const u16* __restrict__ Wkb, const u16* __restrict__ Wvb,
    const float* __restrict__ bq, const float* __restrict__ bk, const float* __restrict__ bv,
    u16* __restrict__ Qr, u16* __restrict__ Kr, u16* __restrict__ Vr) {
  int l = threadIdx.x & 31, w = threadIdx.x >> 5;
  int m = l & 15, h = l >> 4;
  int gw = blockIdx.x * 4 + w;        // 12288 waves total
  int mat = gw >> 12;                 // 0=Q 1=K 2=V (4096 waves each)
  int rem = gw & 4095;
  int b = rem >> 10;
  int it = (rem & 1023) >> 2;         // 256 row tiles
  int cs = rem & 3;                   // 4 col strips of 64
  int i0 = it * 16;
  const u16* W = (mat == 0) ? Wqb : (mat == 1) ? Wkb : Wvb;
  const float* bias = (mat == 0) ? bq : (mat == 1) ? bk : bv;
  u16* Out = (mat == 0) ? Qr : (mat == 1) ? Kr : Vr;

  const u16* Ab = Xr + ((size_t)b * kN + i0) * kC;
  v16bf a[8];
#pragma unroll
  for (int kc = 0; kc < 8; ++kc) a[kc] = load_afrag(Ab, kC, 32 * kc, m, h);

#pragma unroll
  for (int ct = 0; ct < 4; ++ct) {
    int c0 = cs * 64 + ct * 16;
    v8f acc = {};
    const u16* Bt = W + (size_t)c0 * kC;
#pragma unroll
    for (int kc = 0; kc < 8; ++kc)
      acc = wmma_bf16(a[kc], load_bfrag(Bt, kC, 32 * kc, m, h), acc);
    float bv0 = bias[c0 + m];
    u16* Ob = Out + ((size_t)b * kN + i0) * kC + c0 + m;
#pragma unroll
    for (int r = 0; r < 8; ++r)
      Ob[(size_t)(r + 8 * h) * kC] = f2bf(acc[r] + bv0);
  }
}

// ---------------------------------------------------------------- flash attention
// Block = 4 waves, one batch. Waves cooperatively stage the shared K/V j-tiles into
// LDS (async-to-LDS when available); each wave owns 16 query rows with a full d=256
// f32 accumulator. Ar = softmax(Q K^T * s) V + x   (bf16, [B,N,C])
__global__ __launch_bounds__(128, 1) void k_flash(
    const u16* __restrict__ Qr, const u16* __restrict__ Kr, const u16* __restrict__ Vt,
    const u16* __restrict__ Xr, u16* __restrict__ Ar) {
  __shared__ u16 Ks[32 * 256];        // 16 KB: K rows j0..j0+31, row-major [32][256]
  __shared__ u16 Vs[256 * 32];        // 16 KB: Vt rows c=0..255, cols j0..j0+31, [256][32]
  __shared__ u16 Pb[4][16 * 32];      // 4 KB per-wave P staging (D-layout -> A-layout)
  int tid = threadIdx.x;
  int l = tid & 31, w = tid >> 5;
  int m = l & 15, h = l >> 4;
  int b = blockIdx.x >> 6;                              // 64 blocks per batch
  int i0 = (((blockIdx.x & 63) << 2) + w) << 4;         // 16 rows per wave
  const u16* Qb = Qr + ((size_t)b * kN + i0) * kC;
  const u16* Kb = Kr + (size_t)b * kN * kC;
  const u16* Vb = Vt + (size_t)b * kC * kN;
  u16* Pw = &Pb[w][0];

  v8f zero = {};
  v8f O[16];
#pragma unroll
  for (int t = 0; t < 16; ++t) O[t] = zero;
  float mr[8], lr[8];
#pragma unroll
  for (int r = 0; r < 8; ++r) { mr[r] = -3.0e38f; lr[r] = 0.0f; }

  for (int j0 = 0; j0 < kN; j0 += 32) {
    __syncthreads();   // previous iteration's K/V reads complete before restaging
    // ---- cooperative staging: K tile is 16 KB *contiguous* in [N,C] layout ----
    {
      const u16* gk = Kb + (size_t)j0 * kC;
#pragma unroll
      for (int i = 0; i < 8; ++i) {
        int idx = tid + 128 * i;                 // 1024 x 16B = 16 KB
        cp16_to_lds(&Ks[idx * 8], gk + idx * 8);
      }
#pragma unroll
      for (int i = 0; i < 8; ++i) {
        int idx = tid + 128 * i;                 // row c = idx>>2, 16B quarter q = idx&3
        int c = idx >> 2, q = idx & 3;
        cp16_to_lds(&Vs[c * 32 + q * 8], Vb + (size_t)c * kN + j0 + q * 8);
      }
    }
    wait_lds_copies();  // drain this wave's ASYNCcnt before the barrier
    __syncthreads();

    // ---- S = Q K^T (two 16x16 tiles), K from LDS ----
    v8f s0 = zero, s1 = zero;
#pragma unroll
    for (int kc = 0; kc < 8; ++kc) {
      v16bf a = load_afrag(Qb, kC, 32 * kc, m, h);
      s0 = wmma_bf16(a, load_bfrag(Ks, kC, 32 * kc, m, h), s0);
      s1 = wmma_bf16(a, load_bfrag(Ks + 16 * kC, kC, 32 * kc, m, h), s1);
    }
    // ---- online softmax: row M = r + 8h lives across 16 lanes of this half-wave ----
#pragma unroll
    for (int r = 0; r < 8; ++r) {
      float x0 = s0[r] * kScale;
      float x1 = s1[r] * kScale;
      float t = fmaxf(x0, x1);
#pragma unroll
      for (int s = 1; s < 16; s <<= 1) t = fmaxf(t, __shfl_xor(t, s, 32));
      float mn = fmaxf(mr[r], t);
      float alpha = __expf(mr[r] - mn);
      mr[r] = mn;
      float p0 = __expf(x0 - mn);
      float p1 = __expf(x1 - mn);
      float rs = p0 + p1;
#pragma unroll
      for (int s = 1; s < 16; s <<= 1) rs += __shfl_xor(rs, s, 32);
      lr[r] = lr[r] * alpha + rs;
      Pw[(r + 8 * h) * 32 + m] = f2bf(p0);
      Pw[(r + 8 * h) * 32 + 16 + m] = f2bf(p1);
#pragma unroll
      for (int t16 = 0; t16 < 16; ++t16) O[t16][r] *= alpha;
    }
    // ---- O += P * V (P: one 16x32 A-fragment from LDS; V tile from LDS) ----
    v16bf pa = load_afrag(Pw, 32, 0, m, h);
#pragma unroll
    for (int ct = 0; ct < 16; ++ct)
      O[ct] = wmma_bf16(pa, load_bfrag(Vs + ct * 16 * 32, 32, 0, m, h), O[ct]);
  }
  // ---- epilogue: normalize, add residual x, emit bf16 ----
#pragma unroll
  for (int ct = 0; ct < 16; ++ct) {
#pragma unroll
    for (int r = 0; r < 8; ++r) {
      size_t idx = ((size_t)b * kN + i0 + r + 8 * h) * kC + ct * 16 + m;
      float xv = bf2f(Xr[idx]);
      Ar[idx] = f2bf(O[ct][r] / lr[r] + xv);
    }
  }
}

// ---------------------------------------------------------------- output projection
// out[b][c][i] = sum_c' Ar[b][i][c'] * Wo[c][c'] + bo[c]   (store f32, transposed)
__global__ __launch_bounds__(128) void k_oproj(
    const u16* __restrict__ Ar, const u16* __restrict__ Wob, const float* __restrict__ bo,
    float* __restrict__ out) {
  int l = threadIdx.x & 31, w = threadIdx.x >> 5;
  int m = l & 15, h = l >> 4;
  int gw = blockIdx.x * 4 + w;        // 4096 waves total
  int b = gw >> 10;
  int rem = gw & 1023;
  int it = rem >> 2;
  int cs = rem & 3;
  int i0 = it * 16;

  const u16* Ab = Ar + ((size_t)b * kN + i0) * kC;
  v16bf a[8];
#pragma unroll
  for (int kc = 0; kc < 8; ++kc) a[kc] = load_afrag(Ab, kC, 32 * kc, m, h);

#pragma unroll
  for (int ct = 0; ct < 4; ++ct) {
    int c0 = cs * 64 + ct * 16;
    v8f acc = {};
    const u16* Bt = Wob + (size_t)c0 * kC;
#pragma unroll
    for (int kc = 0; kc < 8; ++kc)
      acc = wmma_bf16(a[kc], load_bfrag(Bt, kC, 32 * kc, m, h), acc);
    int c = c0 + m;
    float bias = bo[c];
    float* op = out + ((size_t)b * kC + c) * kN + i0 + 8 * h;
    float4 lo4 = make_float4(acc[0] + bias, acc[1] + bias, acc[2] + bias, acc[3] + bias);
    float4 hi4 = make_float4(acc[4] + bias, acc[5] + bias, acc[6] + bias, acc[7] + bias);
    *(float4*)op = lo4;
    *(float4*)(op + 4) = hi4;
  }
}

// ---------------------------------------------------------------- host launch
extern "C" void kernel_launch(void* const* d_in, const int* in_sizes, int n_in,
                              void* d_out, int out_size, void* d_ws, size_t ws_size,
                              hipStream_t stream) {
  const float* x  = (const float*)d_in[0];
  const float* Wq = (const float*)d_in[1];
  const float* bq = (const float*)d_in[2];
  const float* Wk = (const float*)d_in[3];
  const float* bk = (const float*)d_in[4];
  const float* Wv = (const float*)d_in[5];
  const float* bv = (const float*)d_in[6];
  const float* Wo = (const float*)d_in[7];
  const float* bo = (const float*)d_in[8];
  float* out = (float*)d_out;

  // workspace layout (~51 MB)
  const size_t act = (size_t)kB * kN * kC * sizeof(u16);   // 8 MB per activation buffer
  const size_t wsz = (size_t)kC * kC * sizeof(u16);        // 128 KB per weight
  unsigned char* ws = (unsigned char*)d_ws;
  u16* Xr  = (u16*)(ws + 0 * act);
  u16* Qr  = (u16*)(ws + 1 * act);
  u16* Kr  = (u16*)(ws + 2 * act);
  u16* Vr  = (u16*)(ws + 3 * act);
  u16* Vt  = (u16*)(ws + 4 * act);
  u16* Ar  = (u16*)(ws + 5 * act);
  u16* Wqb = (u16*)(ws + 6 * act + 0 * wsz);
  u16* Wkb = (u16*)(ws + 6 * act + 1 * wsz);
  u16* Wvb = (u16*)(ws + 6 * act + 2 * wsz);
  u16* Wob = (u16*)(ws + 6 * act + 3 * wsz);

  const int wcount = kC * kC;
  k_convert_bf16<<<(wcount + 255) / 256, 256, 0, stream>>>(Wq, Wqb, wcount);
  k_convert_bf16<<<(wcount + 255) / 256, 256, 0, stream>>>(Wk, Wkb, wcount);
  k_convert_bf16<<<(wcount + 255) / 256, 256, 0, stream>>>(Wv, Wvb, wcount);
  k_convert_bf16<<<(wcount + 255) / 256, 256, 0, stream>>>(Wo, Wob, wcount);

  k_pack_x<<<dim3(kN / 32, kC / 32, kB), dim3(32, 8), 0, stream>>>(x, Xr);
  k_qkv<<<3072, 128, 0, stream>>>(Xr, Wqb, Wkb, Wvb, bq, bk, bv, Qr, Kr, Vr);
  k_transpose_v<<<dim3(kN / 32, kC / 32, kB), dim3(32, 8), 0, stream>>>(Vr, Vt);
  k_flash<<<256, 128, 0, stream>>>(Qr, Kr, Vt, Xr, Ar);
  k_oproj<<<1024, 128, 0, stream>>>(Ar, Wob, bo, out);

  (void)in_sizes; (void)n_in; (void)out_size; (void)ws_size;
}
